// Mamba_Layer_55525337203209
// MI455X (gfx1250) — matrix operand order repaired and checked by reading, not compiled, not run on hardware
//
#include <hip/hip_runtime.h>
#include <hip/hip_bf16.h>

// ---------------------------------------------------------------------------
// Bidirectional Mamba layer for gfx1250 (MI455X).
// All GEMMs: pre-converted bf16 operands -> v_wmma_f32_16x16x32_bf16, f32 acc.
// Selective scan: wave32 shuffle-reduced, one thread per (d, n) state.
// B=2, L=2048, DIM=384, D_INNER=768, D_STATE=16, DT_RANK=24 (padded to 32).
// ---------------------------------------------------------------------------

typedef __attribute__((ext_vector_type(16))) __bf16 v16bf;
typedef __attribute__((ext_vector_type(8)))  __bf16 v8bf;
typedef __attribute__((ext_vector_type(8)))  float  v8f;

#define MROWS   4096
#define DIMX    384
#define DINNER  768
#define DSTATE  16
#define LSEQ    2048
#define XDBL_LD 64     // 24 dt + 16 B + 16 C, padded to 64 (f32, for scan)
#define DT_LD   32     // dt operand padded K=24 -> 32 (bf16, for dt GEMM)

// ---------------------------------------------------------------------------
// GEMM: C[M x ldc](f32) = A[M x lda](bf16) * B[N x ldb](bf16)^T (+bias)(+add)
// One wave computes NT adjacent 16x16 tiles, sharing the A fragment.
// Requires KC % 32 == 0 (no masking anywhere).
// MODE 0: plain   MODE 1: softplus(v + bias)   MODE 2: plain + bf16 dt-slice
// ---------------------------------------------------------------------------
template <int MODE, int KC, int NT, int NVALID>
__global__ void gemm_wmma_bf16(const __bf16* __restrict__ A, int lda,
                               const __bf16* __restrict__ B, int ldb,
                               float* __restrict__ C, int ldc,
                               const float* __restrict__ bias,
                               const float* __restrict__ addsrc, int ldadd,
                               __bf16* __restrict__ aux) {
  constexpr int  NTILES = (NVALID + 15) / 16;
  constexpr bool NFULL  = (NVALID % 16) == 0;

  const int lane = threadIdx.x;            // 0..31
  const int half = lane >> 4;              // 0/1
  const int lm   = lane & 15;
  const int tn0  = (blockIdx.y * blockDim.y + threadIdx.y) * NT;
  if (tn0 >= NTILES) return;

  const int m0 = blockIdx.x * 16;
  const __bf16* __restrict__ arow = A + (m0 + lm) * lda;

  int ncol[NT];
  const __bf16* __restrict__ brow[NT];
#pragma unroll
  for (int t = 0; t < NT; ++t) {
    const int n = (tn0 + t) * 16 + lm;
    ncol[t] = n;
    const int n_eff = NFULL ? n : (n < NVALID ? n : NVALID - 1);
    brow[t] = B + n_eff * ldb;
  }

  v8f acc[NT];
#pragma unroll
  for (int t = 0; t < NT; ++t) acc[t] = {};

#pragma unroll
  for (int kc = 0; kc < KC; kc += 32) {
    // A 16x32 bf16 fragment: lane(lm,half): elem j = K[kc+16h+j]
    const v16bf a = *reinterpret_cast<const v16bf*>(arow + kc + half * 16);
    const int kb0 = kc + half * 16;
#pragma unroll
    for (int t = 0; t < NT; ++t) {
      // B 32x16 bf16 fragment: lane(n,half): elem j = B[K = kc+16h+j][n]
      const v16bf b = *reinterpret_cast<const v16bf*>(brow[t] + kb0);
      acc[t] = __builtin_amdgcn_wmma_f32_16x16x32_bf16(
          false, a, false, b, (short)0, acc[t], false, false);
    }
  }

  const int mbase = m0 + half * 8;         // C/D: VGPR r -> M = r + 8*half
#pragma unroll
  for (int t = 0; t < NT; ++t) {
    const int n = ncol[t];
    if (NFULL || n < NVALID) {
      const float bv = bias ? bias[n] : 0.f;
#pragma unroll
      for (int r = 0; r < 8; ++r) {
        float v = acc[t][r] + bv;
        if (addsrc) v += addsrc[(mbase + r) * ldadd + n];
        if (MODE == 1) v = (v > 20.f) ? v : log1pf(__expf(v));  // softplus
        C[(mbase + r) * ldc + n] = v;
        if (MODE == 2 && n < DT_LD) {      // zero-padded bf16 dt operand
          aux[(mbase + r) * DT_LD + n] = (n < 24) ? (__bf16)v : (__bf16)0.f;
        }
      }
    }
  }
}

// f32 -> bf16 elementwise convert (weights / activations), n % 1 arbitrary.
__global__ void cvt_f32_bf16_kernel(const float* __restrict__ src,
                                    __bf16* __restrict__ dst, int n) {
  const int i = blockIdx.x * blockDim.x + threadIdx.x;
  if (i < n) dst[i] = (__bf16)src[i];
}

// dt_proj_w (768 x 24) f32 -> (768 x 32) bf16, zero-padded K.
__global__ void cvt_pad_dtw_kernel(const float* __restrict__ src,
                                   __bf16* __restrict__ dst) {
  const int i = blockIdx.x * blockDim.x + threadIdx.x;   // 768*32
  const int r = i >> 5, c = i & 31;
  dst[i] = (c < 24) ? (__bf16)src[r * 24 + c] : (__bf16)0.f;
}

// Row-wise LayerNorm + ReLU over 384 features; bf16 output.  One wave per row.
__global__ void ln_relu_kernel(const float* __restrict__ h,
                               const float* __restrict__ g,
                               const float* __restrict__ bta,
                               __bf16* __restrict__ out) {
  const int row  = blockIdx.x;
  const int lane = threadIdx.x;
  const float* __restrict__ hp = h + row * DIMX;
  float vals[12];
  float s = 0.f, s2 = 0.f;
#pragma unroll
  for (int i = 0; i < 12; ++i) {
    float v = hp[lane + 32 * i];
    vals[i] = v; s += v; s2 += v * v;
  }
#pragma unroll
  for (int m = 1; m < 32; m <<= 1) {
    s  += __shfl_xor(s,  m, 32);
    s2 += __shfl_xor(s2, m, 32);
  }
  const float mu  = s * (1.f / DIMX);
  const float var = s2 * (1.f / DIMX) - mu * mu;
  const float inv = rsqrtf(var + 1e-5f);
  __bf16* __restrict__ op = out + row * DIMX;
#pragma unroll
  for (int i = 0; i < 12; ++i) {
    const int c = lane + 32 * i;
    float v = (vals[i] - mu) * inv * g[c] + bta[c];
    op[c] = (__bf16)(v > 0.f ? v : 0.f);
  }
}

// Depthwise width-4 conv (causal fwd / anti-causal bwd) + bias + SiLU.
// xz: (4096 x 1536) row-major; channels [0,768).  Writes f32 (scan) + bf16 (GEMM).
__global__ void dwconv_silu_kernel(const float* __restrict__ xz,
                                   const float* __restrict__ w,
                                   const float* __restrict__ cb,
                                   float* __restrict__ xcf,
                                   __bf16* __restrict__ xcb, int dir) {
  const int gid = blockIdx.x * blockDim.x + threadIdx.x;
  if (gid >= MROWS * DINNER) return;
  const int m = gid / DINNER;
  const int d = gid - m * DINNER;
  const int b = m >> 11;          // L = 2048
  const int l = m & (LSEQ - 1);
  float acc = cb[d];
#pragma unroll
  for (int j = 0; j < 4; ++j) {
    const int ls = dir ? (l + 3 - j) : (l - 3 + j);
    if (ls >= 0 && ls < LSEQ)
      acc += w[d * 4 + j] * xz[((b << 11) + ls) * (2 * DINNER) + d];
  }
  const float s = acc / (1.f + __expf(-acc));  // SiLU
  xcf[gid] = s;
  xcb[gid] = (__bf16)s;
}

// Selective scan.  One thread per (d, n) state; 16 n-lanes butterfly-reduce.
// blockDim = 256 (16 d-channels per block), grid = (768/16, B).
// y = (scan + xc*D) * silu(z);  accumulate==1 adds into y (backward dir).
__global__ void selective_scan_kernel(const float* __restrict__ delta,
                                      const float* __restrict__ xc,
                                      const float* __restrict__ xdbl,
                                      const float* __restrict__ xz,
                                      const float* __restrict__ A_log,
                                      const float* __restrict__ Dp,
                                      float* __restrict__ y,
                                      int dir, int accumulate) {
  const int tid  = threadIdx.x;
  const int nidx = tid & 15;
  const int d    = blockIdx.x * 16 + (tid >> 4);
  const int b    = blockIdx.y;
  const float Adn = -__expf(A_log[d * DSTATE + nidx]);  // A = -exp(A_log)
  const float Dd  = Dp[d];
  float h = 0.f;
  const int baseRow = b * LSEQ;
  for (int step = 0; step < LSEQ; ++step) {
    const int l   = dir ? (LSEQ - 1 - step) : step;
    const int row = baseRow + l;
    const float dl = delta[row * DINNER + d];
    const float xv = xc[row * DINNER + d];
    const float Bv = xdbl[row * XDBL_LD + 24 + nidx];
    const float Cv = xdbl[row * XDBL_LD + 40 + nidx];
    const float dA = __expf(dl * Adn);
    h = dA * h + (dl * xv) * Bv;
    float ys = h * Cv;
    ys += __shfl_xor(ys, 1, 32);
    ys += __shfl_xor(ys, 2, 32);
    ys += __shfl_xor(ys, 4, 32);
    ys += __shfl_xor(ys, 8, 32);
    if (nidx == 0) {
      const float zv = xz[row * (2 * DINNER) + DINNER + d];
      const float gate = zv / (1.f + __expf(-zv));       // silu(z)
      const float val = (ys + xv * Dd) * gate;
      if (accumulate) y[row * DINNER + d] += val;
      else            y[row * DINNER + d]  = val;
    }
  }
}

static inline int cdiv(int a, int b) { return (a + b - 1) / b; }

extern "C" void kernel_launch(void* const* d_in, const int* in_sizes, int n_in,
                              void* d_out, int out_size, void* d_ws, size_t ws_size,
                              hipStream_t stream) {
  (void)in_sizes; (void)n_in; (void)out_size; (void)ws_size;
  const float* x         = (const float*)d_in[0];   // (B,L,384)
  const float* lin_w     = (const float*)d_in[1];   // (384,384)
  const float* lin_b     = (const float*)d_in[2];
  const float* ln_g      = (const float*)d_in[3];
  const float* ln_b      = (const float*)d_in[4];
  const float* in_proj_w = (const float*)d_in[5];   // (1536,384)
  const float* conv_w    = (const float*)d_in[6];   // (768,1,4)
  const float* conv_b    = (const float*)d_in[7];
  const float* x_proj_w  = (const float*)d_in[8];   // (56,768)
  const float* dt_proj_w = (const float*)d_in[9];   // (768,24)
  const float* dt_proj_b = (const float*)d_in[10];
  const float* A_log     = (const float*)d_in[11];  // (768,16)
  const float* A_b_log   = (const float*)d_in[12];
  const float* Dp        = (const float*)d_in[13];
  const float* out_proj_w= (const float*)d_in[14];  // (384,768)
  float* out = (float*)d_out;

  // ---- workspace carve: f32 region then bf16 region (all 32B aligned) ----
  float* ws    = (float*)d_ws;
  float* h_pre = ws;                               // 4096*384
  float* xzbuf = h_pre + MROWS * DIMX;             // 4096*1536
  float* xcbuf = xzbuf + MROWS * 2 * DINNER;       // 4096*768
  float* xdbl  = xcbuf + MROWS * DINNER;           // 4096*64
  float* dlt   = xdbl  + MROWS * XDBL_LD;          // 4096*768
  float* ybuf  = dlt   + MROWS * DINNER;           // 4096*768

  __bf16* bfp     = (__bf16*)(ybuf + MROWS * DINNER);
  __bf16* x_bf    = bfp;             bfp += MROWS * DIMX;
  __bf16* hln_bf  = bfp;             bfp += MROWS * DIMX;
  __bf16* xc_bf   = bfp;             bfp += MROWS * DINNER;
  __bf16* dt_bf   = bfp;             bfp += MROWS * DT_LD;
  __bf16* y_bf    = bfp;             bfp += MROWS * DINNER;
  __bf16* wlin_bf = bfp;             bfp += DIMX * DIMX;
  __bf16* winp_bf = bfp;             bfp += 2 * DINNER * DIMX;
  __bf16* wxp_bf  = bfp;             bfp += 56 * DINNER;
  __bf16* wdt_bf  = bfp;             bfp += DINNER * DT_LD;
  __bf16* wout_bf = bfp;             bfp += DIMX * DINNER;

  const dim3 gb4(32, 4);
  const dim3 gb1(32, 1);

  // ---- one-time (per launch) bf16 conversions: weights + input x ----
  cvt_f32_bf16_kernel<<<cdiv(MROWS * DIMX, 256), 256, 0, stream>>>(
      x, x_bf, MROWS * DIMX);
  cvt_f32_bf16_kernel<<<cdiv(DIMX * DIMX, 256), 256, 0, stream>>>(
      lin_w, wlin_bf, DIMX * DIMX);
  cvt_f32_bf16_kernel<<<cdiv(2 * DINNER * DIMX, 256), 256, 0, stream>>>(
      in_proj_w, winp_bf, 2 * DINNER * DIMX);
  cvt_f32_bf16_kernel<<<cdiv(56 * DINNER, 256), 256, 0, stream>>>(
      x_proj_w, wxp_bf, 56 * DINNER);
  cvt_f32_bf16_kernel<<<cdiv(DIMX * DINNER, 256), 256, 0, stream>>>(
      out_proj_w, wout_bf, DIMX * DINNER);
  cvt_pad_dtw_kernel<<<cdiv(DINNER * DT_LD, 256), 256, 0, stream>>>(
      dt_proj_w, wdt_bf);

  // 1) h = x @ lin_w^T + lin_b            M=4096, N=384, K=384, NT=2
  gemm_wmma_bf16<0, 384, 2, 384><<<dim3(MROWS / 16, 3), gb4, 0, stream>>>(
      x_bf, DIMX, wlin_bf, DIMX, h_pre, DIMX, lin_b, nullptr, 0, nullptr);
  // 2) LayerNorm + ReLU -> bf16
  ln_relu_kernel<<<MROWS, 32, 0, stream>>>(h_pre, ln_g, ln_b, hln_bf);
  // 3) xz = h @ in_proj_w^T               N=1536, K=384, NT=4
  gemm_wmma_bf16<0, 384, 4, 1536><<<dim3(MROWS / 16, 6), gb4, 0, stream>>>(
      hln_bf, DIMX, winp_bf, DIMX, xzbuf, 2 * DINNER, nullptr, nullptr, 0,
      nullptr);

  for (int dir = 0; dir < 2; ++dir) {
    // 4) depthwise conv + SiLU (causal fwd / anti-causal bwd)
    dwconv_silu_kernel<<<(MROWS * DINNER) / 256, 256, 0, stream>>>(
        xzbuf, conv_w, conv_b, xcbuf, xc_bf, dir);
    // 5) x_dbl = xc @ x_proj_w^T   N=56 (4 tiles, NT=4), K=768;
    //    MODE 2 also emits zero-padded bf16 dt operand (4096 x 32)
    gemm_wmma_bf16<2, 768, 4, 56><<<dim3(MROWS / 16, 1), gb1, 0, stream>>>(
        xc_bf, DINNER, wxp_bf, DINNER, xdbl, XDBL_LD, nullptr, nullptr, 0,
        dt_bf);
    // 6) delta = softplus(dt @ dt_proj_w^T + dt_proj_b)   N=768, K=32, NT=2
    gemm_wmma_bf16<1, 32, 2, 768><<<dim3(MROWS / 16, 6), gb4, 0, stream>>>(
        dt_bf, DT_LD, wdt_bf, DT_LD, dlt, DINNER, dt_proj_b, nullptr, 0,
        nullptr);
    // 7) selective scan (+ D skip + silu(z) gate); bwd accumulates
    selective_scan_kernel<<<dim3(DINNER / 16, 2), 256, 0, stream>>>(
        dlt, xcbuf, xdbl, xzbuf, dir ? A_b_log : A_log, Dp, ybuf, dir, dir);
  }

  // y -> bf16 for the output projection
  cvt_f32_bf16_kernel<<<cdiv(MROWS * DINNER, 256), 256, 0, stream>>>(
      ybuf, y_bf, MROWS * DINNER);
  // 8) out = y @ out_proj_w^T + x_skip    N=384, K=768, NT=2
  gemm_wmma_bf16<0, 768, 2, 384><<<dim3(MROWS / 16, 3), gb4, 0, stream>>>(
      y_bf, DINNER, wout_bf, DINNER, out, DIMX, nullptr, x, DIMX, nullptr);
}
